// LSTM_3143916061177
// MI455X (gfx1250) — matrix operand order repaired and checked by reading, not compiled, not run on hardware
//
#include <hip/hip_runtime.h>

// ---------------------------------------------------------------------------
// LSTM for MI455X (gfx1250): bf16 WMMA GEMMs + persistent recurrent kernel.
//   B=64, T=512, I=1024, H=1024, 4H=4096
// Phase 1: x_proj[32768,4096] = x@Wx + b  -- 2x4-tile WMMA GEMM, ping-pong
//          K pipeline (no fragment copies, no WAR hazard NOPs, no spills)
// Phase 2: persistent kernel (32 blocks), 512 steps:
//          h tile staged to LDS via Tensor Data Mover (TDM), A-frags via ds,
//          Wh bf16 (8MB) L2-resident, c lives in VGPRs, device-wide barrier.
// ---------------------------------------------------------------------------

typedef __attribute__((ext_vector_type(16))) __bf16       v16bf;
typedef __attribute__((ext_vector_type(8)))  __bf16       v8bf;
typedef __attribute__((ext_vector_type(8)))  float        v8f;
typedef __attribute__((ext_vector_type(4)))  unsigned int u32x4;
typedef __attribute__((ext_vector_type(4)))  int          i32x4;
typedef __attribute__((ext_vector_type(8)))  int          i32x8;

#define B_  64
#define T_  512
#define I_  1024
#define H_  1024
#define G4  4096        // 4*H
#define R_  32768       // B*T
#define NBLK_STEP 32    // persistent-kernel blocks (all resident)

#if defined(__has_builtin)
#  if __has_builtin(__builtin_amdgcn_tensor_load_to_lds)
#    define TDM_AVAIL 1
#  endif
#  if __has_builtin(__builtin_amdgcn_s_wait_tensorcnt)
#    define TWAIT_AVAIL 1
#  endif
#endif
#ifndef TDM_AVAIL
#  define TDM_AVAIL 0
#endif
#ifndef TWAIT_AVAIL
#  define TWAIT_AVAIL 0
#endif

// ---------------- workspace layout (bytes) ----------------
// packWx : [256 nt][32 kt][32 lane][16 j] bf16   =   8 MiB
// packWh : same                                  =   8 MiB
// packX  : [2048 rt][32 kt][32 lane][16 j] bf16  =  64 MiB
// xproj  : [32768][4096] f32                     = 512 MiB
// hfrag  : [2][mt=4][ht=64][16 row][16 col] bf16 = 256 KiB
// barrier: 2 x u32
static const size_t SZ_PW   = (size_t)I_ * G4 * 2;
static const size_t OFF_PWX = 0;
static const size_t OFF_PWH = OFF_PWX + SZ_PW;
static const size_t OFF_PX  = OFF_PWH + SZ_PW;
static const size_t SZ_PX   = (size_t)R_ * I_ * 2;
static const size_t OFF_XP  = OFF_PX + SZ_PX;
static const size_t SZ_XP   = (size_t)R_ * G4 * 4;
static const size_t OFF_HB  = OFF_XP + SZ_XP;
static const size_t SZ_HB   = (size_t)2 * 4 * 64 * 256 * 2;   // 256 KiB
static const size_t OFF_BAR = OFF_HB + SZ_HB;

// ---------------- pack W (fp32 [1024][4096]) -> WMMA B-fragment bf16 -------
// B 32x16 bf16 tile: lane = 16*((k_in>>3)&1) + n_in, j = (k_in&7) | ((k_in>>4)<<3)
__global__ void lstm_pack_w_bfrag(const float* __restrict__ W,
                                  __bf16* __restrict__ dst) {
  size_t tid  = (size_t)blockIdx.x * blockDim.x + threadIdx.x;  // I_*G4
  size_t j    = tid & 15;
  size_t lane = (tid >> 4) & 31;
  size_t kt   = (tid >> 9) & 31;
  size_t nt   = tid >> 14;
  size_t k = kt * 32 + (j & 7) + (lane >> 4) * 8 + (j >> 3) * 16;
  size_t n = nt * 16 + (lane & 15);
  dst[tid] = (__bf16)W[k * G4 + n];
}

// ---------------- pack x (fp32 [R][1024]) -> WMMA A-fragment bf16 ----------
// A 16x32 bf16 tile: lane = m + 16*((k>>3)&1), j = (k&7) | ((k>>4)<<3)
__global__ void lstm_pack_x_afrag(const float* __restrict__ X,
                                  __bf16* __restrict__ dst) {
  size_t tid  = (size_t)blockIdx.x * blockDim.x + threadIdx.x;  // R_*I_
  size_t j    = tid & 15;
  size_t lane = (tid >> 4) & 31;
  size_t kt   = (tid >> 9) & 31;
  size_t rt   = tid >> 14;
  size_t m    = lane & 15;
  size_t hi   = lane >> 4;
  size_t kin  = (j & 7) + 8 * hi + 16 * (j >> 3);
  dst[tid] = (__bf16)X[(rt * 16 + m) * I_ + kt * 32 + kin];
}

// ---------------- init: zero h0 (tile layout) + barrier state --------------
__global__ void lstm_init(unsigned* __restrict__ h_u32,
                          unsigned* __restrict__ bar) {
  size_t tid = (size_t)blockIdx.x * blockDim.x + threadIdx.x;
  if (tid < (size_t)4 * 64 * 256 * 2 / 4) h_u32[tid] = 0u;  // hfrag[0] = 0
  if (tid == 0) { bar[0] = 0u; bar[1] = 0u; }
}

// ---------------- phase 1: x_proj = packX @ packWx + bias ------------------
// 256 thr (8 waves); each wave owns a 32x64 output macro-tile (2x4 WMMA
// tiles). Ping-pong K pipeline: even K-steps compute from buffer 0 while
// buffer 1 loads, and vice versa -- no fragment copies, loads overlap WMMAs.
__global__ void __launch_bounds__(256)
lstm_xproj_gemm(const __bf16* __restrict__ pX,
                const __bf16* __restrict__ pW,
                const float*  __restrict__ bias,
                float* __restrict__ xproj) {
  const int lane = threadIdx.x & 31;
  const int wave = threadIdx.x >> 5;
  const int rt0  = blockIdx.y * 2;                   // 2 M-tiles
  const int nt0  = (blockIdx.x * 8 + wave) * 4;      // 4 N-tiles

  v8f acc[2][4] = {};
  v16bf a0[2], b0[4], a1[2], b1[4];

  auto ldA = [&](v16bf* d, int ktv) {
#pragma unroll
    for (int mi = 0; mi < 2; ++mi)
      d[mi] = *(const v16bf*)(pX + (((size_t)(rt0 + mi) * 32 + ktv) * 32 + lane) * 16);
  };
  auto ldB = [&](v16bf* d, int ktv) {
#pragma unroll
    for (int ni = 0; ni < 4; ++ni)
      d[ni] = *(const v16bf*)(pW + (((size_t)(nt0 + ni) * 32 + ktv) * 32 + lane) * 16);
  };
  auto mm = [&](const v16bf* a, const v16bf* b) {
#pragma unroll
    for (int mi = 0; mi < 2; ++mi)
#pragma unroll
      for (int ni = 0; ni < 4; ++ni)
        acc[mi][ni] = __builtin_amdgcn_wmma_f32_16x16x32_bf16(
            false, a[mi], false, b[ni], (short)0, acc[mi][ni], false, false);
  };

  ldA(a0, 0);
  ldB(b0, 0);
  for (int kt = 0; kt < 32; kt += 2) {
    ldA(a1, kt + 1);              // fill buf1 while buf0 computes
    ldB(b1, kt + 1);
    mm(a0, b0);
    if (kt + 2 < 32) {            // fill buf0 while buf1 computes
      ldA(a0, kt + 2);
      ldB(b0, kt + 2);
    }
    mm(a1, b1);
  }

  // C/D layout: VGPR r, lane l -> row = r + 8*(l/16), col = l%16
  const int ncol = lane & 15;
#pragma unroll
  for (int mi = 0; mi < 2; ++mi) {
    const int rbase = (rt0 + mi) * 16 + (lane >> 4) * 8;
#pragma unroll
    for (int ni = 0; ni < 4; ++ni) {
      const int col = (nt0 + ni) * 16 + ncol;
      const float bv = bias[col];
#pragma unroll
      for (int r = 0; r < 8; ++r)
        xproj[(size_t)(rbase + r) * G4 + col] = acc[mi][ni][r] + bv;
    }
  }
}

// ---------------- TDM: contiguous 1-D global -> LDS copy -------------------
__device__ __forceinline__ void tdm_load_1d(unsigned lds_off, const void* gptr,
                                            unsigned ndw) {
#if TDM_AVAIL && defined(__HIP_DEVICE_COMPILE__)
  unsigned long long ga = (unsigned long long)(uintptr_t)gptr;
  u32x4 g0;
  g0.x = 1u;                                             // count=1, user mode
  g0.y = lds_off;                                        // LDS byte address
  g0.z = (unsigned)ga;                                   // global addr lo
  g0.w = (unsigned)((ga >> 32) & 0x01ffffffull) | (2u << 30);  // hi | type=2
  i32x8 g1;
  g1[0] = (int)(2u << 16);                               // data_size = 4B
  g1[1] = (int)((ndw & 0xffffu) << 16);                  // tensor_dim0[15:0]
  g1[2] = (int)(((ndw >> 16) & 0xffffu) | (1u << 16));   // dim0 hi | dim1=1
  g1[3] = (int)((ndw & 0xffffu) << 16);                  // tile_dim0
  g1[4] = 1;                                             // tile_dim1 = 1
  g1[5] = (int)ndw;                                      // dim0_stride lo
  g1[6] = (int)((ndw >> 16) & 0xffffu);                  // dim0_stride hi
  g1[7] = 0;
  i32x4 z4 = {0, 0, 0, 0};
#if __clang_major__ >= 23
  i32x8 z8 = {0, 0, 0, 0, 0, 0, 0, 0};
  __builtin_amdgcn_tensor_load_to_lds(g0, g1, z4, z4, z8, 0);
#else
  __builtin_amdgcn_tensor_load_to_lds(g0, g1, z4, z4, 0);
#endif
#if TWAIT_AVAIL
  __builtin_amdgcn_s_wait_tensorcnt(0);
#endif
#else
  (void)lds_off; (void)gptr; (void)ndw;
#endif
}

// ---------------- device-wide barrier (persistent kernel) ------------------
__device__ __forceinline__ void grid_sync(unsigned* cnt, unsigned* gen) {
  __threadfence();            // release my h stores to L2
  __syncthreads();
  if (threadIdx.x == 0) {
    unsigned g = __hip_atomic_load(gen, __ATOMIC_RELAXED, __HIP_MEMORY_SCOPE_AGENT);
    if (__hip_atomic_fetch_add(cnt, 1u, __ATOMIC_ACQ_REL,
                               __HIP_MEMORY_SCOPE_AGENT) == NBLK_STEP - 1) {
      __hip_atomic_store(cnt, 0u, __ATOMIC_RELAXED, __HIP_MEMORY_SCOPE_AGENT);
      __hip_atomic_fetch_add(gen, 1u, __ATOMIC_RELEASE, __HIP_MEMORY_SCOPE_AGENT);
    } else {
      while (__hip_atomic_load(gen, __ATOMIC_ACQUIRE,
                               __HIP_MEMORY_SCOPE_AGENT) == g)
        __builtin_amdgcn_s_sleep(2);
    }
  }
  __syncthreads();
  __threadfence();            // acquire: drop stale near-cache lines
}

// ---------------- phase 2: persistent recurrence ---------------------------
// 32 blocks x 8 waves. Block owns batch tile mt = blockIdx>>3 (16 rows);
// its 8 waves cover 8 consecutive h-column tiles x 4 gates.
// Per step: TDM stages the block's 32KB h tile (16x16-tile layout) into LDS
// once (8x reuse), A-frags come from LDS (2 aligned b128 ds loads), B-frags
// ping-pong from L2-resident packWh, 128 WMMAs/wave, c in VGPRs.
__global__ void __launch_bounds__(256)
lstm_steps(const __bf16* __restrict__ pWh,
           const float*  __restrict__ xproj,
           __bf16* __restrict__ hfrag,
           float*  __restrict__ out,
           unsigned* __restrict__ bar) {
  __shared__ __bf16 ldsH[16 * H_];          // 32 KiB: this mt's h tiles
  const int lane = threadIdx.x & 31;
  const int wave = threadIdx.x >> 5;
  const int mt   = blockIdx.x >> 3;         // 0..3
  const int ht   = (blockIdx.x & 7) * 8 + wave;   // 0..63
  unsigned* cnt = bar;
  unsigned* gen = bar + 1;

  v8f c = {};                               // cell state in VGPRs, all 512 t
  const int rbase = mt * 16 + (lane >> 4) * 8;
  const int ncol  = lane & 15;
  const int hcol  = ht * 16 + ncol;
  // A-frag from LDS tile layout [ht2][16 row][16 col]:
  // lane L: lo = k-tile's even col-tile, row L%16, cols 8*(L/16)..+8
  //         hi = odd col-tile (+512B). Both 16B aligned.
  const unsigned lds_lo = (unsigned)((lane & 15) * 32 + (lane >> 4) * 16);
  const unsigned lds_base = (unsigned)(uintptr_t)&ldsH[0];

  auto ldA = [&](int ktv) -> v16bf {
    union { v16bf v; struct { v8bf lo, hi; } s; } a;
    const char* lp = (const char*)ldsH + ktv * 1024 + lds_lo;
    a.s.lo = *(const v8bf*)lp;
    a.s.hi = *(const v8bf*)(lp + 512);
    return a.v;
  };
  auto ldB = [&](v16bf* d, int ktv) {
#pragma unroll
    for (int g = 0; g < 4; ++g)
      d[g] = *(const v16bf*)(pWh + (((size_t)(g * 64 + ht) * 32 + ktv) * 32 + lane) * 16);
  };

  int cur = 0;
  for (int t = 0; t < T_; ++t) {
    const __bf16* hsrc = hfrag + ((size_t)cur * 4 + mt) * (64 * 256);
#if TDM_AVAIL && defined(__HIP_DEVICE_COMPILE__)
    if (wave == 0)
      tdm_load_1d(lds_base, hsrc, (16 * H_ * 2) / 4);   // 8192 dwords
#else
    {   // cooperative fallback: 2048 x 16B
      const uint4* s = (const uint4*)hsrc;
      uint4* d = (uint4*)&ldsH[0];
      for (int i = threadIdx.x; i < (16 * H_ * 2) / 16; i += 256) d[i] = s[i];
    }
#endif
    __syncthreads();

    v8f acc[4] = {};
    v16bf b0[4], b1[4];
    ldB(b0, 0);
    for (int kt = 0; kt < 32; kt += 2) {
      v16bf ae = ldA(kt);
      ldB(b1, kt + 1);                        // fill buf1 while buf0 computes
#pragma unroll
      for (int g = 0; g < 4; ++g)
        acc[g] = __builtin_amdgcn_wmma_f32_16x16x32_bf16(
            false, ae, false, b0[g], (short)0, acc[g], false, false);
      v16bf ao = ldA(kt + 1);
      if (kt + 2 < 32) ldB(b0, kt + 2);       // fill buf0 while buf1 computes
#pragma unroll
      for (int g = 0; g < 4; ++g)
        acc[g] = __builtin_amdgcn_wmma_f32_16x16x32_bf16(
            false, ao, false, b1[g], (short)0, acc[g], false, false);
    }

    // ---- cell update: gates = acc + x_proj, fp32 activations ----
    v8f hv;
#pragma unroll
    for (int r = 0; r < 8; ++r) {
      const int row = rbase + r;
      const size_t xr = ((size_t)row * T_ + t) * G4;
      float gi = acc[0][r] + xproj[xr + 0 * H_ + hcol];
      float gf = acc[1][r] + xproj[xr + 1 * H_ + hcol];
      float gg = acc[2][r] + xproj[xr + 2 * H_ + hcol];
      float go = acc[3][r] + xproj[xr + 3 * H_ + hcol];
      float iv = 1.0f / (1.0f + __expf(-gi));
      float fv = 1.0f / (1.0f + __expf(-gf));
      float gv = tanhf(gg);
      float ov = 1.0f / (1.0f + __expf(-go));
      float cv = fv * c[r] + iv * gv;
      c[r] = cv;
      float h = ov * tanhf(cv);
      hv[r] = h;
      out[((size_t)row * T_ + t) * H_ + hcol] = h;
    }
    // store bf16 h into next buffer, 16x16-tile layout [mt][ht][row][col]
    __bf16* hdst = hfrag + ((size_t)(cur ^ 1) * 4 + mt) * (64 * 256) + (size_t)ht * 256;
#pragma unroll
    for (int r = 0; r < 8; ++r)
      hdst[((lane >> 4) * 8 + r) * 16 + ncol] = (__bf16)hv[r];

    grid_sync(cnt, gen);    // also fences LDS reuse for next step's staging
    cur ^= 1;
  }
}

// ---------------------------------------------------------------------------
extern "C" void kernel_launch(void* const* d_in, const int* in_sizes, int n_in,
                              void* d_out, int out_size, void* d_ws, size_t ws_size,
                              hipStream_t stream) {
  (void)in_sizes; (void)n_in; (void)out_size; (void)ws_size;
  const float* x  = (const float*)d_in[0];   // [64,512,1024]
  const float* Wx = (const float*)d_in[1];   // [1024,4096]
  const float* Wh = (const float*)d_in[2];   // [1024,4096]
  const float* bb = (const float*)d_in[3];   // [4096]
  float* out = (float*)d_out;                // [64,512,1024]

  char* ws = (char*)d_ws;
  __bf16* pWx   = (__bf16*)(ws + OFF_PWX);
  __bf16* pWh   = (__bf16*)(ws + OFF_PWH);
  __bf16* pX    = (__bf16*)(ws + OFF_PX);
  float*  xproj = (float*) (ws + OFF_XP);
  __bf16* hfrag = (__bf16*)(ws + OFF_HB);
  unsigned* bar = (unsigned*)(ws + OFF_BAR);

  lstm_pack_w_bfrag<<<(I_ * (size_t)G4) / 256, 256, 0, stream>>>(Wx, pWx);
  lstm_pack_w_bfrag<<<(I_ * (size_t)G4) / 256, 256, 0, stream>>>(Wh, pWh);
  lstm_pack_x_afrag<<<((size_t)R_ * I_) / 256, 256, 0, stream>>>(x, pX);
  lstm_init<<<128, 256, 0, stream>>>((unsigned*)hfrag, bar);

  lstm_xproj_gemm<<<dim3(8, R_ / 32), 256, 0, stream>>>(pX, pWx, bb, xproj);
  lstm_steps<<<NBLK_STEP, 256, 0, stream>>>(pWh, xproj, hfrag, out, bar);
}
